// RNN8Class_33474975105476
// MI455X (gfx1250) — matrix operand order repaired and checked by reading, not compiled, not run on hardware
//
#include <hip/hip_runtime.h>
#include <stdint.h>

// ---------------------------------------------------------------------------
// CDNA5 / gfx1250 WMMA types (wave32)
// ---------------------------------------------------------------------------
typedef __attribute__((ext_vector_type(16))) _Float16 v16h;
typedef __attribute__((ext_vector_type(8)))  float    v8f;

union FragU {
  v16h v;
  _Float16 e[16];
  uint32_t u[8];
};

__device__ __forceinline__ float dev_tanh(float x) {
#if __has_builtin(__builtin_amdgcn_tanhf)
  return __builtin_amdgcn_tanhf(x);      // v_tanh_f32 on gfx1250
#else
  return tanhf(x);
#endif
}

// Build WMMA A-fragments (16x32 f16 per fragment) from a row-major f32
// weight matrix W[*, K].  Contiguous K order (matches B fragments):
//   lane<16 : M=lane, slot s holds K pairs {2s,2s+1}
//   lane>=16: same M, K offset +16
template<int K>
__device__ __forceinline__ void load_A_frags(const float* __restrict__ W,
                                             int row, int lane, FragU* A,
                                             bool valid) {
#pragma unroll
  for (int kb = 0; kb < K / 32; ++kb) {
#pragma unroll
    for (int s = 0; s < 8; ++s) {
      int kk = 32 * kb + 2 * s + ((lane < 16) ? 0 : 16);
      float2 wv;
      if (valid) {
        wv = *(const float2*)(W + (size_t)row * K + kk);
      } else {
        wv.x = 0.0f; wv.y = 0.0f;
      }
      A[kb].e[2 * s]     = (_Float16)wv.x;
      A[kb].e[2 * s + 1] = (_Float16)wv.y;
    }
  }
}

// ---------------------------------------------------------------------------
// Input-projection GEMM (transposed orientation):  Y[bt, 0:128] = X[bt,0:K]*W^T + b1 + b2
//   D(16H x 16bt) = A(W rows, 16x32 f16) x B(X^T, 32x16 f16) accumulated f32.
// Workgroup = 8 waves; wave w owns H rows 16w..16w+15; block-strided over bt tiles.
// X tile (16 rows x K) is staged through LDS (exactly 4KB for f32/K=64 and f16/K=128).
// ---------------------------------------------------------------------------
template<typename TIN, int K>
__global__ __launch_bounds__(256) void proj_kernel(
    const TIN* __restrict__ X, const float* __restrict__ W,
    const float* __restrict__ b1, const float* __restrict__ b2,
    float* __restrict__ Y, int Mtiles) {
  constexpr int NF = K / 32;
  __shared__ __align__(16) uint4 tile4[256];  // 4KB staging
  const int tid  = threadIdx.x;
  const int lane = tid & 31;
  const int w    = tid >> 5;
  const int n    = lane & 15;
  const int hi   = (lane < 16) ? 0 : 8;

  FragU A[NF];
  load_A_frags<K>(W, 16 * w + n, lane, A, true);

  const int Hb = 16 * w + hi;
  float4 bA = *(const float4*)(b1 + Hb);
  float4 bB = *(const float4*)(b2 + Hb);
  float4 bC = *(const float4*)(b1 + Hb + 4);
  float4 bD = *(const float4*)(b2 + Hb + 4);
  float bias[8] = {bA.x + bB.x, bA.y + bB.y, bA.z + bB.z, bA.w + bB.w,
                   bC.x + bD.x, bC.y + bD.y, bC.z + bD.z, bC.w + bD.w};

  for (int tt = blockIdx.x; tt < Mtiles; tt += gridDim.x) {
    // cooperative stage of the 16xK input tile (contiguous rows)
    const uint4* src = (const uint4*)(X + (size_t)tt * 16 * K);
    tile4[tid] = src[tid];
    __syncthreads();

    v8f acc = {};
#pragma unroll
    for (int kb = 0; kb < NF; ++kb) {
      FragU Bf;
      if constexpr (sizeof(TIN) == 4) {
        const float* rowp =
            ((const float*)tile4) + n * K + 32 * kb + ((lane < 16) ? 0 : 16);
#pragma unroll
        for (int q = 0; q < 4; ++q) {
          float4 f = *(const float4*)(rowp + 4 * q);
          Bf.e[4 * q + 0] = (_Float16)f.x;
          Bf.e[4 * q + 1] = (_Float16)f.y;
          Bf.e[4 * q + 2] = (_Float16)f.z;
          Bf.e[4 * q + 3] = (_Float16)f.w;
        }
      } else {
        const _Float16* rowp =
            ((const _Float16*)tile4) + n * K + 32 * kb + ((lane < 16) ? 0 : 16);
        uint4 h0 = *(const uint4*)(rowp);
        uint4 h1 = *(const uint4*)(rowp + 8);
        Bf.u[0] = h0.x; Bf.u[1] = h0.y; Bf.u[2] = h0.z; Bf.u[3] = h0.w;
        Bf.u[4] = h1.x; Bf.u[5] = h1.y; Bf.u[6] = h1.z; Bf.u[7] = h1.w;
      }
      acc = __builtin_amdgcn_wmma_f32_16x16x32_f16(
          false, A[kb].v, false, Bf.v, (short)0, acc, false, false);
    }

    float* outp = Y + (size_t)(16 * tt + n) * 128 + Hb;
    float4 o0 = {acc[0] + bias[0], acc[1] + bias[1], acc[2] + bias[2], acc[3] + bias[3]};
    float4 o1 = {acc[4] + bias[4], acc[5] + bias[5], acc[6] + bias[6], acc[7] + bias[7]};
    *(float4*)outp       = o0;
    *(float4*)(outp + 4) = o1;
    __syncthreads();
  }
}

// ---------------------------------------------------------------------------
// Recurrence: h_t = tanh(xp_t + h_{t-1} W_hh^T), computed transposed:
//   D(16H x 16batch) = A(W_hh rows, resident VGPRs) x B(h^T from LDS)
// 4 blocks x 256 threads; block owns 16 batch rows, wave w owns H rows 16w..+15.
// Latency-critical: two independent 2-deep WMMA accumulation chains per step
// (instead of one 4-deep chain) to halve the XDL dependency depth; per step:
// 4 WMMA, tanh(v_tanh_f32), pack f16, 1 global_store_b128 (h out),
// 1 ds_store_b128 + barrier + 8 ds_load_b128 (state exchange, double-buffered).
// ---------------------------------------------------------------------------
__global__ __launch_bounds__(256) void rec_kernel(
    const float* __restrict__ XP,   // [B][T][128] f32 (input projection)
    const float* __restrict__ Whh,  // [128][128] f32
    _Float16* __restrict__ Hout,    // [B][T][128] f16
    int T) {
  constexpr int H = 128;
  __shared__ __align__(16) uint32_t hx[2][4][32][8];  // 8KB double buffer
  const int tid  = threadIdx.x;
  const int lane = tid & 31;
  const int w    = tid >> 5;
  const int n    = lane & 15;
  const int hi   = (lane < 16) ? 0 : 8;
  const int batch = blockIdx.x * 16 + n;

  FragU A[4];
  load_A_frags<128>(Whh, 16 * w + n, lane, A, true);

  FragU Bf[4];
#pragma unroll
  for (int kb = 0; kb < 4; ++kb)
#pragma unroll
    for (int q = 0; q < 8; ++q) Bf[kb].u[q] = 0u;  // h0 = 0

  // D-tile -> next B-fragment routing through LDS:
  // wave w produces K rows 16w..16w+15 of frag kb=w/2 (half = w&1)
  const int kbw   = w >> 1;
  const int dlane = n + ((w & 1) ? 16 : 0);
  const int dslot = (lane < 16) ? 0 : 4;

  const float* xp = XP + ((size_t)batch * T) * H + 16 * w + hi;
  _Float16*    hp = Hout + ((size_t)batch * T) * H + 16 * w + hi;

  for (int t = 0; t < T; ++t) {
    float4 xa = *(const float4*)xp;
    float4 xb = *(const float4*)(xp + 4);
    xp += H;

    // two independent accumulation chains (halved dependency depth)
    v8f acc0 = {};
    v8f acc1 = {};
    acc0 = __builtin_amdgcn_wmma_f32_16x16x32_f16(
        false, A[0].v, false, Bf[0].v, (short)0, acc0, false, false);
    acc1 = __builtin_amdgcn_wmma_f32_16x16x32_f16(
        false, A[1].v, false, Bf[1].v, (short)0, acc1, false, false);
    acc0 = __builtin_amdgcn_wmma_f32_16x16x32_f16(
        false, A[2].v, false, Bf[2].v, (short)0, acc0, false, false);
    acc1 = __builtin_amdgcn_wmma_f32_16x16x32_f16(
        false, A[3].v, false, Bf[3].v, (short)0, acc1, false, false);

    float pre[8];
    pre[0] = (acc0[0] + xa.x) + acc1[0];
    pre[1] = (acc0[1] + xa.y) + acc1[1];
    pre[2] = (acc0[2] + xa.z) + acc1[2];
    pre[3] = (acc0[3] + xa.w) + acc1[3];
    pre[4] = (acc0[4] + xb.x) + acc1[4];
    pre[5] = (acc0[5] + xb.y) + acc1[5];
    pre[6] = (acc0[6] + xb.z) + acc1[6];
    pre[7] = (acc0[7] + xb.w) + acc1[7];

    float hv[8];
#pragma unroll
    for (int r = 0; r < 8; ++r) hv[r] = dev_tanh(pre[r]);

    uint32_t pk[4];
#pragma unroll
    for (int j = 0; j < 4; ++j) {
      union { uint32_t u; _Float16 e[2]; } p;
      p.e[0] = (_Float16)hv[2 * j];
      p.e[1] = (_Float16)hv[2 * j + 1];
      pk[j] = p.u;
    }
    uint4 pv;
    pv.x = pk[0]; pv.y = pk[1]; pv.z = pk[2]; pv.w = pk[3];

    *(uint4*)hp = pv;   // h_t to global as f16 (contiguous 8 H values)
    hp += H;

    *(uint4*)&hx[t & 1][kbw][dlane][dslot] = pv;  // exchange
    __syncthreads();
#pragma unroll
    for (int kb = 0; kb < 4; ++kb) {
      const uint32_t* s = &hx[t & 1][kb][lane][0];
      uint4 a0 = *(const uint4*)s;
      uint4 a1 = *(const uint4*)(s + 4);
      Bf[kb].u[0] = a0.x; Bf[kb].u[1] = a0.y; Bf[kb].u[2] = a0.z; Bf[kb].u[3] = a0.w;
      Bf[kb].u[4] = a1.x; Bf[kb].u[5] = a1.y; Bf[kb].u[6] = a1.z; Bf[kb].u[7] = a1.w;
    }
  }
}

// ---------------------------------------------------------------------------
// FC head:  Out[bt, 0:8] = H2[bt,0:128] * W_fc^T + b_fc  (C=8 padded to 16 rows)
// One wave per 16-bt tile; A = W_fc (rows >= 8 zero); B read directly from global.
// ---------------------------------------------------------------------------
__global__ __launch_bounds__(256) void fc_kernel(
    const _Float16* __restrict__ H2,  // [BT][128] f16
    const float* __restrict__ Wfc,    // [8][128]
    const float* __restrict__ bfc,    // [8]
    float* __restrict__ Out,          // [BT][8]
    int Mtiles) {
  const int tid  = threadIdx.x;
  const int lane = tid & 31;
  const int w    = tid >> 5;
  const int n    = lane & 15;
  const int nwaves = gridDim.x * 8;
  const int wid    = blockIdx.x * 8 + w;

  FragU A[4];
  load_A_frags<128>(Wfc, n, lane, A, n < 8);

  float4 b0 = *(const float4*)bfc;
  float4 b1 = *(const float4*)(bfc + 4);

  for (int tt = wid; tt < Mtiles; tt += nwaves) {
    const _Float16* base =
        H2 + (size_t)(16 * tt + n) * 128 + ((lane < 16) ? 0 : 16);
    v8f acc = {};
#pragma unroll
    for (int kb = 0; kb < 4; ++kb) {
      const _Float16* p = base + 32 * kb;
      uint4 h0 = *(const uint4*)p;
      uint4 h1 = *(const uint4*)(p + 8);
      FragU Bf;
      Bf.u[0] = h0.x; Bf.u[1] = h0.y; Bf.u[2] = h0.z; Bf.u[3] = h0.w;
      Bf.u[4] = h1.x; Bf.u[5] = h1.y; Bf.u[6] = h1.z; Bf.u[7] = h1.w;
      acc = __builtin_amdgcn_wmma_f32_16x16x32_f16(
          false, A[kb].v, false, Bf.v, (short)0, acc, false, false);
    }
    if (lane < 16) {  // rows M=0..7 (the 8 valid classes) live in lanes 0-15
      float* op = Out + (size_t)(16 * tt + n) * 8;
      float4 o0 = {acc[0] + b0.x, acc[1] + b0.y, acc[2] + b0.z, acc[3] + b0.w};
      float4 o1 = {acc[4] + b1.x, acc[5] + b1.y, acc[6] + b1.z, acc[7] + b1.w};
      *(float4*)op       = o0;
      *(float4*)(op + 4) = o1;
    }
  }
}

// ---------------------------------------------------------------------------
// Launch: proj0 -> rec0 -> proj1 -> rec1 -> fc  (all on stream, ws reused)
// ---------------------------------------------------------------------------
extern "C" void kernel_launch(void* const* d_in, const int* in_sizes, int n_in,
                              void* d_out, int out_size, void* d_ws, size_t ws_size,
                              hipStream_t stream) {
  const int B = 64, T = 2048, H = 128;
  const size_t BT = (size_t)B * T;
  const int Mtiles = (int)(BT / 16);  // 8192

  const float* x     = (const float*)d_in[0];
  const float* W_ih0 = (const float*)d_in[1];
  const float* W_hh0 = (const float*)d_in[2];
  const float* b_ih0 = (const float*)d_in[3];
  const float* b_hh0 = (const float*)d_in[4];
  const float* W_ih1 = (const float*)d_in[5];
  const float* W_hh1 = (const float*)d_in[6];
  const float* b_ih1 = (const float*)d_in[7];
  const float* b_hh1 = (const float*)d_in[8];
  const float* W_fc  = (const float*)d_in[9];
  const float* b_fc  = (const float*)d_in[10];

  float*    xp   = (float*)d_ws;                                       // BT*128 f32
  _Float16* hbuf = (_Float16*)((char*)d_ws + BT * H * sizeof(float));  // BT*128 f16

  // layer 0 input projection: x (f32, K=64) -> xp
  proj_kernel<float, 64><<<512, 256, 0, stream>>>(x, W_ih0, b_ih0, b_hh0, xp, Mtiles);
  // layer 0 recurrence -> hbuf (f16)
  rec_kernel<<<4, 256, 0, stream>>>(xp, W_hh0, hbuf, T);
  // layer 1 input projection: hbuf (f16, K=128) -> xp (xp0 is dead, reuse)
  proj_kernel<_Float16, 128><<<512, 256, 0, stream>>>(hbuf, W_ih1, b_ih1, b_hh1, xp, Mtiles);
  // layer 1 recurrence -> hbuf (h1 is dead, reuse)
  rec_kernel<<<4, 256, 0, stream>>>(xp, W_hh1, hbuf, T);
  // classifier head -> d_out
  fc_kernel<<<256, 256, 0, stream>>>(hbuf, W_fc, b_fc, (float*)d_out, Mtiles);
}
